// SSMLayer_55997783605677
// MI455X (gfx1250) — matrix operand order repaired and checked by reading, not compiled, not run on hardware
//
#include <hip/hip_runtime.h>
#include <hip/hip_bf16.h>

// Problem constants (match reference setup_inputs)
#define DMODEL 1024
#define NSTATE 128
#define BATCH  8
#define TLEN   4096
#define MROWS  (BATCH * TLEN)   // 32768 rows when u viewed as [M, D]
#define CHUNKS 64
#define CLEN   (TLEN / CHUNKS)  // 64 timesteps per chunk
#define KSLAB  128              // K-panel of B_w staged in LDS per iteration

typedef __attribute__((ext_vector_type(16))) _Float16 v16h;
typedef __attribute__((ext_vector_type(8)))  _Float16 v8h;
typedef __attribute__((ext_vector_type(8)))  float    v8f;
typedef __attribute__((ext_vector_type(4)))  int      v4i;

// ---------------------------------------------------------------------------
// Async global->LDS copy (CDNA5 GLOBAL_LOAD_ASYNC_TO_LDS_B128, ASYNCcnt
// tracked). Builtin signature (from hipcc diagnostic): v4i AS(1)* src,
// v4i AS(3)* dst, imm offset, imm cpol. Guarded fallback keeps compilation
// safe if the builtin is absent.
// ---------------------------------------------------------------------------
__device__ __forceinline__ void async_copy16(const _Float16* g, _Float16* l) {
#if defined(__gfx1250__) && __has_builtin(__builtin_amdgcn_global_load_async_to_lds_b128)
    __builtin_amdgcn_global_load_async_to_lds_b128(
        (__attribute__((address_space(1))) v4i*)(const void*)g,
        (__attribute__((address_space(3))) v4i*)(void*)l, 0, 0);
#else
    *(v8h*)l = *(const v8h*)g;
#endif
}

__device__ __forceinline__ void async_wait_all() {
#if defined(__gfx1250__) && __has_builtin(__builtin_amdgcn_s_wait_asynccnt)
    __builtin_amdgcn_s_wait_asynccnt(0);
#elif defined(__gfx1250__)
    asm volatile("s_wait_asynccnt 0" ::: "memory");
#endif
}

// ---------------------------------------------------------------------------
// Kernel 0: one-time fp32 -> f16 conversion of the weight matrices.
// ---------------------------------------------------------------------------
__global__ void wconvert(const float* __restrict__ Bw, const float* __restrict__ Cw,
                         _Float16* __restrict__ Bwh, _Float16* __restrict__ Cwh) {
    int i = blockIdx.x * blockDim.x + threadIdx.x;
    if (i < NSTATE * DMODEL) {
        Bwh[i] = (_Float16)Bw[i];
        Cwh[i] = (_Float16)Cw[i];
    }
}

// ---------------------------------------------------------------------------
// Kernel 1: Bu[m, n] = (sum_d u[m,d] * B_w[n,d] + B_b[n]) * dt[n]
//
// Block = 8 waves; wave w owns row tile (blockIdx*8 + w), all 8 N-tiles.
// B_w is staged K-slab by K-slab into LDS (async copy), shared by all 8
// waves; the WMMA loop then feeds from ds_load_b128 instead of waiting on
// VMEM round trips. Row stride is padded to 136 halfs (68 dwords => bank
// stride 4) so the 16-lane fragment reads are bank-conflict free.
// ---------------------------------------------------------------------------
__global__ __launch_bounds__(256) void gemm_bu(
        const float* __restrict__ u, const _Float16* __restrict__ Bwh,
        const float* __restrict__ Bb, const float* __restrict__ log_dt,
        float* __restrict__ Bu) {
    __shared__ __align__(16) _Float16 Bs[NSTATE][KSLAB + 8];   // ~34 KB

    const int tid     = threadIdx.x;
    const int lane    = tid & 31;
    const int wave    = tid >> 5;
    const int rowbase = (blockIdx.x * 8 + wave) * 16;
    const int r16     = lane & 15;
    const int hi      = lane >> 4;   // lane-half select
    const int klo     = hi * 8;      // A-fragment K octet offset
    const int koff    = hi * 16;     // unused; B reads also use klo

    v8f acc[8];
    v8f zero = {};
#pragma unroll
    for (int t = 0; t < 8; ++t) acc[t] = zero;

    const float* urow = u + (size_t)(rowbase + r16) * DMODEL;

    for (int s = 0; s < DMODEL / KSLAB; ++s) {
        const int k0s = s * KSLAB;

        // Cooperative async stage of the 128 x 128 f16 B panel (32 KB).
        // 2048 16-byte chunks, 8 per thread.
#pragma unroll
        for (int i = 0; i < 8; ++i) {
            int chunk = tid + i * 256;
            int row   = chunk >> 4;          // KSLAB/8 = 16 chunks per row
            int colc  = (chunk & 15) * 8;
            async_copy16(Bwh + (size_t)row * DMODEL + k0s + colc, &Bs[row][colc]);
        }
        async_wait_all();
        __syncthreads();

        // Prefetch next slab's u panel while we compute this one.
        if (k0s + KSLAB < DMODEL) __builtin_prefetch(urow + k0s + KSLAB + klo, 0, 0);

#pragma unroll
        for (int kk = 0; kk < KSLAB; kk += 32) {
            const int k0 = k0s + kk;
            float4 f0 = *(const float4*)(urow + k0 + klo);
            float4 f1 = *(const float4*)(urow + k0 + klo + 4);
            float4 f2 = *(const float4*)(urow + k0 + 16 + klo);
            float4 f3 = *(const float4*)(urow + k0 + 16 + klo + 4);
            v16h a;
            a[0]  = (_Float16)f0.x; a[1]  = (_Float16)f0.y; a[2]  = (_Float16)f0.z; a[3]  = (_Float16)f0.w;
            a[4]  = (_Float16)f1.x; a[5]  = (_Float16)f1.y; a[6]  = (_Float16)f1.z; a[7]  = (_Float16)f1.w;
            a[8]  = (_Float16)f2.x; a[9]  = (_Float16)f2.y; a[10] = (_Float16)f2.z; a[11] = (_Float16)f2.w;
            a[12] = (_Float16)f3.x; a[13] = (_Float16)f3.y; a[14] = (_Float16)f3.z; a[15] = (_Float16)f3.w;

            // Explicit fragment array: forces distinct registers so the 16
            // ds_load_b128 issue as a pipelined clause, not one-at-a-time.
            v16h bfr[8];
#pragma unroll
            for (int t = 0; t < 8; ++t) {
                const _Float16* bp = &Bs[t * 16 + r16][kk + klo];
                v8h b0 = *(const v8h*)bp;
                v8h b1 = *(const v8h*)(bp + 16);
                bfr[t] = __builtin_shufflevector(b0, b1, 0,1,2,3,4,5,6,7,8,9,10,11,12,13,14,15);
            }
#pragma unroll
            for (int t = 0; t < 8; ++t) {
                acc[t] = __builtin_amdgcn_wmma_f32_16x16x32_f16(
                    false, a, false, bfr[t], (short)0, acc[t], false, false);
            }
        }
        __syncthreads();
    }

    // Epilogue: fold bias + dt scaling. C/D layout: VGPR v, lane l ->
    // row = (l<16 ? v : v+8), col = l&15.
#pragma unroll
    for (int t = 0; t < 8; ++t) {
        int   ncol = t * 16 + r16;
        float dtv  = fminf(__expf(log_dt[ncol]), 1.0f);
        float bb   = Bb[ncol];
#pragma unroll
        for (int v = 0; v < 8; ++v) {
            int row = rowbase + klo + v;
            Bu[(size_t)row * NSTATE + ncol] = (acc[t][v] + bb) * dtv;
        }
    }
}

// ---------------------------------------------------------------------------
// Scan phase A: per-(b, chunk, n) local scan with zero init, in place over Bu.
// ---------------------------------------------------------------------------
__global__ void scan_local(const float* __restrict__ log_A, const float* __restrict__ log_dt,
                           float* __restrict__ Bu, float* __restrict__ localfinal) {
    const int n = threadIdx.x;
    const int b = blockIdx.x / CHUNKS;
    const int c = blockIdx.x % CHUNKS;
    const float dt = fminf(__expf(log_dt[n]), 1.0f);
    const float Ab = __expf(-__expf(log_A[n]) * dt);
    size_t base = ((size_t)(b * TLEN + c * CLEN)) * NSTATE + n;
    float h = 0.0f;
#pragma unroll 4
    for (int j = 0; j < CLEN; ++j) {
        float x = Bu[base + (size_t)j * NSTATE];
        h = Ab * h + x;
        Bu[base + (size_t)j * NSTATE] = h;
    }
    localfinal[((size_t)b * CHUNKS + c) * NSTATE + n] = h;
}

// ---------------------------------------------------------------------------
// Scan phase B: serial combine across the 64 chunks per (b, n) sequence.
// ---------------------------------------------------------------------------
__global__ void scan_carry(const float* __restrict__ h0, const float* __restrict__ log_A,
                           const float* __restrict__ log_dt,
                           const float* __restrict__ localfinal, float* __restrict__ carry) {
    const int n = threadIdx.x;
    const int b = blockIdx.x;
    const float dt  = fminf(__expf(log_dt[n]), 1.0f);
    const float AbL = __expf(-__expf(log_A[n]) * dt * (float)CLEN);
    float cv = h0[b * NSTATE + n];
    for (int c = 0; c < CHUNKS; ++c) {
        size_t idx = ((size_t)b * CHUNKS + c) * NSTATE + n;
        carry[idx] = cv;
        cv = AbL * cv + localfinal[idx];
    }
}

// ---------------------------------------------------------------------------
// Scan phase C: h[s+j] = local[j] + A_bar^(j+1) * carry; emit hs as f16.
// ---------------------------------------------------------------------------
__global__ void scan_apply(const float* __restrict__ log_A, const float* __restrict__ log_dt,
                           const float* __restrict__ Bu, const float* __restrict__ carry,
                           _Float16* __restrict__ hs_h) {
    const int n = threadIdx.x;
    const int b = blockIdx.x / CHUNKS;
    const int c = blockIdx.x % CHUNKS;
    const float dt = fminf(__expf(log_dt[n]), 1.0f);
    const float Ab = __expf(-__expf(log_A[n]) * dt);
    const float cv = carry[((size_t)b * CHUNKS + c) * NSTATE + n];
    size_t base = ((size_t)(b * TLEN + c * CLEN)) * NSTATE + n;
    float p = Ab;
#pragma unroll 4
    for (int j = 0; j < CLEN; ++j) {
        float val = Bu[base + (size_t)j * NSTATE] + p * cv;
        p *= Ab;
        hs_h[base + (size_t)j * NSTATE] = (_Float16)val;
    }
}

// ---------------------------------------------------------------------------
// Kernel 3: y[m, d] = sum_n hs[m,n] * C_w[d,n] + C_b[d] + D[d] * u[m,d]
// One block (8 waves) per 16-row tile; wave w owns columns [w*128, w*128+128).
// All 4 A-fragments preloaded; B fragments loaded into an explicit array per
// K-step so the 16 global_load_b128 pipeline ahead of the 8 WMMAs.
// ---------------------------------------------------------------------------
__global__ __launch_bounds__(256) void gemm_y(
        const _Float16* __restrict__ hs_h, const _Float16* __restrict__ Cwh,
        const float* __restrict__ Cb, const float* __restrict__ Dp,
        const float* __restrict__ u, float* __restrict__ y) {
    const int lane    = threadIdx.x & 31;
    const int wave    = threadIdx.x >> 5;
    const int rowbase = blockIdx.x * 16;
    const int dbase   = wave * 128;
    const int r16     = lane & 15;
    const int hi      = lane >> 4;
    const int klo     = hi * 8;
    const int koff    = hi * 16;

    v8f acc[8];
    v8f zero = {};
#pragma unroll
    for (int t = 0; t < 8; ++t) acc[t] = zero;

    const _Float16* hrow = hs_h + (size_t)(rowbase + r16) * NSTATE;

    // Preload all 4 A-fragments (K = 128) in one go.
    v16h afr[4];
#pragma unroll
    for (int ks = 0; ks < 4; ++ks) {
        const int k0 = ks * 32;
        v8h a0 = *(const v8h*)(hrow + k0 + klo);
        v8h a1 = *(const v8h*)(hrow + k0 + 16 + klo);
        afr[ks] = __builtin_shufflevector(a0, a1, 0,1,2,3,4,5,6,7,8,9,10,11,12,13,14,15);
    }

#pragma unroll
    for (int ks = 0; ks < 4; ++ks) {
        const int k0 = ks * 32;
        v16h bfr[8];
#pragma unroll
        for (int t = 0; t < 8; ++t) {
            const _Float16* cp = Cwh + (size_t)(dbase + t * 16 + r16) * NSTATE + k0 + klo;
            v8h b0 = *(const v8h*)cp;
            v8h b1 = *(const v8h*)(cp + 16);
            bfr[t] = __builtin_shufflevector(b0, b1, 0,1,2,3,4,5,6,7,8,9,10,11,12,13,14,15);
        }
#pragma unroll
        for (int t = 0; t < 8; ++t) {
            acc[t] = __builtin_amdgcn_wmma_f32_16x16x32_f16(
                false, afr[ks], false, bfr[t], (short)0, acc[t], false, false);
        }
    }

#pragma unroll
    for (int t = 0; t < 8; ++t) {
        int   col = dbase + t * 16 + r16;
        float cb  = Cb[col];
        float dp  = Dp[col];
#pragma unroll
        for (int v = 0; v < 8; ++v) {
            int row = rowbase + klo + v;
            size_t idx = (size_t)row * DMODEL + col;
            y[idx] = acc[t][v] + cb + dp * u[idx];
        }
    }
}

// ---------------------------------------------------------------------------
// Host launcher. Inputs (setup_inputs order):
// 0:u 1:h0 2:log_A 3:B_w 4:B_b 5:C_w 6:C_b 7:D_param 8:log_dt
// ---------------------------------------------------------------------------
extern "C" void kernel_launch(void* const* d_in, const int* in_sizes, int n_in,
                              void* d_out, int out_size, void* d_ws, size_t ws_size,
                              hipStream_t stream) {
    const float* u      = (const float*)d_in[0];
    const float* h0     = (const float*)d_in[1];
    const float* log_A  = (const float*)d_in[2];
    const float* B_w    = (const float*)d_in[3];
    const float* B_b    = (const float*)d_in[4];
    const float* C_w    = (const float*)d_in[5];
    const float* C_b    = (const float*)d_in[6];
    const float* D_p    = (const float*)d_in[7];
    const float* log_dt = (const float*)d_in[8];
    float* y = (float*)d_out;

    // Workspace carve-up (~25.5 MB total).
    char* w = (char*)d_ws;
    float* Bu = (float*)w;                 w += (size_t)MROWS * NSTATE * sizeof(float);      // 16 MB
    float* localfinal = (float*)w;         w += (size_t)BATCH * CHUNKS * NSTATE * sizeof(float);
    float* carry = (float*)w;              w += (size_t)BATCH * CHUNKS * NSTATE * sizeof(float);
    _Float16* hs_h = (_Float16*)w;         w += (size_t)MROWS * NSTATE * sizeof(_Float16);   // 8 MB
    _Float16* Bwh = (_Float16*)w;          w += (size_t)NSTATE * DMODEL * sizeof(_Float16);
    _Float16* Cwh = (_Float16*)w;          w += (size_t)DMODEL * NSTATE * sizeof(_Float16);
    (void)in_sizes; (void)n_in; (void)out_size; (void)ws_size;

    wconvert<<<(NSTATE * DMODEL + 255) / 256, 256, 0, stream>>>(B_w, C_w, Bwh, Cwh);
    gemm_bu<<<MROWS / (16 * 8), 256, 0, stream>>>(u, Bwh, B_b, log_dt, Bu);
    scan_local<<<BATCH * CHUNKS, NSTATE, 0, stream>>>(log_A, log_dt, Bu, localfinal);
    scan_carry<<<BATCH, NSTATE, 0, stream>>>(h0, log_A, log_dt, localfinal, carry);
    scan_apply<<<BATCH * CHUNKS, NSTATE, 0, stream>>>(log_A, log_dt, Bu, carry, hs_h);
    gemm_y<<<MROWS / 16, 256, 0, stream>>>(hs_h, Cwh, C_b, D_p, u, y);
}
